// AttributedGAT_75668733820971
// MI455X (gfx1250) — compile-verified, matrix operand order checked by reading, not run on hardware
//
#include <hip/hip_runtime.h>
#include <hip/hip_bf16.h>
#include <math.h>

typedef __attribute__((ext_vector_type(16))) _Float16 v16h;
typedef __attribute__((ext_vector_type(8)))  float    v8f;

#define IN_C  128
#define HID_C 64
#define EDGE_D 32
#define NEG_SLOPE 0.2f

static inline int cdiv(long long a, long long b) { return (int)((a + b - 1) / b); }

// ---------------------------------------------------------------------------
// small utility kernels
// ---------------------------------------------------------------------------
__global__ void fill_kernel(float* __restrict__ p, float v, long long n) {
    long long i = (long long)blockIdx.x * blockDim.x + threadIdx.x;
    if (i < n) p[i] = v;
}

// v1[k] = sum_c We1[k,c]*att_edge1[c]; v2 likewise (both We are 32x64)
__global__ void ve_kernel(const float* __restrict__ We1, const float* __restrict__ ae1,
                          const float* __restrict__ We2, const float* __restrict__ ae2,
                          float* __restrict__ v1, float* __restrict__ v2) {
    int t = threadIdx.x;                 // 64 threads
    const float* We = (t < 32) ? We1 : We2;
    const float* at = (t < 32) ? ae1 : ae2;
    float*       v  = (t < 32) ? v1  : v2;
    int k = t & 31;
    float s = 0.f;
    #pragma unroll 8
    for (int c = 0; c < 64; ++c) s += We[k * 64 + c] * at[c];
    v[k] = s;
}

// per-edge: ae = edge_attr . v  (both layers), plus scalar segment sums for
// the self-loop "mean" attrs
__global__ void edge_ae_kernel(const float* __restrict__ ea,
                               const int* __restrict__ dst,
                               const float* __restrict__ v1, const float* __restrict__ v2,
                               float* __restrict__ ae1, float* __restrict__ ae2,
                               float* __restrict__ sae1, float* __restrict__ sae2,
                               float* __restrict__ cnt, int E_) {
    int e = blockIdx.x * blockDim.x + threadIdx.x;
    if (e >= E_) return;
    const float4* a4 = (const float4*)(ea + (long long)e * EDGE_D);
    float d1 = 0.f, d2 = 0.f;
    #pragma unroll
    for (int q = 0; q < EDGE_D / 4; ++q) {
        float4 av = a4[q];
        d1 += av.x * v1[q*4+0] + av.y * v1[q*4+1] + av.z * v1[q*4+2] + av.w * v1[q*4+3];
        d2 += av.x * v2[q*4+0] + av.y * v2[q*4+1] + av.z * v2[q*4+2] + av.w * v2[q*4+3];
    }
    ae1[e] = d1; ae2[e] = d2;
    int d = dst[e];
    atomicAdd(sae1 + d, d1);
    atomicAdd(sae2 + d, d2);
    atomicAdd(cnt + d, 1.0f);
}

// ---------------------------------------------------------------------------
// Pack W[K,64] (f32) into f16 WMMA B-fragment order:
// Wp[((kt*4 + ct)*32 + lane)*16 + j] = (f16) W[(kt*32 + koff(lane,j))*64 + ct*16 + (lane&15)]
// koff: lanes 0-15 -> {0..7,16..23}; lanes 16-31 -> {8..15,24..31}
// ---------------------------------------------------------------------------
template<int K>
__global__ void pack_w_kernel(const float* __restrict__ W, _Float16* __restrict__ Wp) {
    int t = blockIdx.x * blockDim.x + threadIdx.x;   // one thread per (kt,ct,lane)
    const int nfrag = (K / 32) * 4 * 32;
    if (t >= nfrag) return;
    int lane = t & 31;
    int ct   = (t >> 5) & 3;
    int kt   = t >> 7;
    int g = lane >> 4;
    int col = ct * 16 + (lane & 15);
    _Float16* o = Wp + (long long)t * 16;
    #pragma unroll
    for (int j = 0; j < 8; ++j) {
        o[j]     = (_Float16)W[(kt * 32 + g * 8 + j) * 64 + col];
        o[j + 8] = (_Float16)W[(kt * 32 + 16 + g * 8 + j) * 64 + col];
    }
}

// ---------------------------------------------------------------------------
// WMMA GEMM: H[nrows,64] = X[nrows,K] @ W[K,64]   (f16 mul, f32 acc)
// one wave per 16-row tile; 4 column tiles of 16; B pre-packed in fragment order
// ---------------------------------------------------------------------------
template<int K>
__global__ void gemm_wmma(const float* __restrict__ X, const _Float16* __restrict__ Wp,
                          float* __restrict__ H, int nrows) {
    const int wave = (blockIdx.x * blockDim.x + threadIdx.x) >> 5;
    const int lane = threadIdx.x & 31;
    const int row0 = wave * 16;
    if (row0 >= nrows) return;
    const int g = lane >> 4;     // half-wave group (K-split per ISA layout)
    const int r = lane & 15;     // row (A) / col (B) within tile

    v8f acc0 = {}, acc1 = {}, acc2 = {}, acc3 = {};

    int arow = row0 + r; if (arow >= nrows) arow = nrows - 1;
    const float4* xr4 = (const float4*)(X + (long long)arow * K);
    const v16h*   Wpv = (const v16h*)Wp;

    #pragma unroll
    for (int kt = 0; kt < K / 32; ++kt) {
        const int kk = kt * 32;
        // A fragment (16x32 f16): two contiguous 8-float runs per lane -> 4x b128
        float4 p0 = xr4[(kk + g * 8) >> 2];
        float4 p1 = xr4[((kk + g * 8) >> 2) + 1];
        float4 p2 = xr4[(kk + 16 + g * 8) >> 2];
        float4 p3 = xr4[((kk + 16 + g * 8) >> 2) + 1];
        v16h a;
        a[0]  = (_Float16)p0.x; a[1]  = (_Float16)p0.y; a[2]  = (_Float16)p0.z; a[3]  = (_Float16)p0.w;
        a[4]  = (_Float16)p1.x; a[5]  = (_Float16)p1.y; a[6]  = (_Float16)p1.z; a[7]  = (_Float16)p1.w;
        a[8]  = (_Float16)p2.x; a[9]  = (_Float16)p2.y; a[10] = (_Float16)p2.z; a[11] = (_Float16)p2.w;
        a[12] = (_Float16)p3.x; a[13] = (_Float16)p3.y; a[14] = (_Float16)p3.z; a[15] = (_Float16)p3.w;
        // B fragments: one contiguous 32-byte load each
        v16h b0 = Wpv[(kt * 4 + 0) * 32 + lane];
        v16h b1 = Wpv[(kt * 4 + 1) * 32 + lane];
        v16h b2 = Wpv[(kt * 4 + 2) * 32 + lane];
        v16h b3 = Wpv[(kt * 4 + 3) * 32 + lane];
        acc0 = __builtin_amdgcn_wmma_f32_16x16x32_f16(false, a, false, b0, (short)0, acc0, false, false);
        acc1 = __builtin_amdgcn_wmma_f32_16x16x32_f16(false, a, false, b1, (short)0, acc1, false, false);
        acc2 = __builtin_amdgcn_wmma_f32_16x16x32_f16(false, a, false, b2, (short)0, acc2, false, false);
        acc3 = __builtin_amdgcn_wmma_f32_16x16x32_f16(false, a, false, b3, (short)0, acc3, false, false);
    }
    // D layout: VGPR v -> row = v + 8*g, col = r (per 16-col tile)
    #pragma unroll
    for (int ct = 0; ct < 4; ++ct) {
        const v8f accv = (ct == 0) ? acc0 : (ct == 1) ? acc1 : (ct == 2) ? acc2 : acc3;
        #pragma unroll
        for (int v = 0; v < 8; ++v) {
            int row = row0 + v + 8 * g;
            if (row < nrows) H[(long long)row * 64 + ct * 16 + r] = accv[v];
        }
    }
}

// per-node attention scalars: as[n] = h[n].att_src, ad[n] = h[n].att_dst
__global__ void node_attn_kernel(const float* __restrict__ h,
                                 const float* __restrict__ asv, const float* __restrict__ adv,
                                 float* __restrict__ as_, float* __restrict__ ad_, int N_) {
    int n = blockIdx.x * blockDim.x + threadIdx.x;
    if (n >= N_) return;
    const float4* hn = (const float4*)(h + (long long)n * 64);
    float s = 0.f, t = 0.f;
    #pragma unroll
    for (int q = 0; q < 16; ++q) {
        float4 hv = hn[q];
        s += hv.x * asv[q*4+0] + hv.y * asv[q*4+1] + hv.z * asv[q*4+2] + hv.w * asv[q*4+3];
        t += hv.x * adv[q*4+0] + hv.y * adv[q*4+1] + hv.z * adv[q*4+2] + hv.w * adv[q*4+3];
    }
    as_[n] = s; ad_[n] = t;
}

__device__ inline void atomicMaxFloat(float* addr, float val) {
    // standard int-punned float max (correct for all orderings incl. -inf init)
    if (val >= 0.f) atomicMax((int*)addr, __float_as_int(val));
    else            atomicMin((unsigned int*)addr, __float_as_uint(val));
}

// pass 1 over E real edges + N self-loops: alpha = leaky_relu(...), segment max
__global__ void alpha_max_kernel(const int* __restrict__ src, const int* __restrict__ dst,
                                 const float* __restrict__ ae, const float* __restrict__ sae,
                                 const float* __restrict__ cnt,
                                 const float* __restrict__ as_, const float* __restrict__ ad_,
                                 float* __restrict__ alpha, float* __restrict__ m,
                                 int E_, int N_) {
    int i = blockIdx.x * blockDim.x + threadIdx.x;
    if (i >= E_ + N_) return;
    int s, d; float aev;
    if (i < E_) { s = src[i]; d = dst[i]; aev = ae[i]; }
    else { s = d = i - E_; aev = sae[d] / fmaxf(cnt[d], 1.0f); }  // self-loop, mean attr
    float a = as_[s] + ad_[d] + aev;
    a = (a > 0.f) ? a : a * NEG_SLOPE;
    alpha[i] = a;
    atomicMaxFloat(m + d, a);
}

// pass 2: p = exp(alpha - m[dst]); denom[dst] += p; acc[dst] += p*h[src]
// one wave per edge, 2 columns per lane (gathers & atomics stay L2-resident)
__global__ void scatter_kernel(const int* __restrict__ src, const int* __restrict__ dst,
                               const float* __restrict__ alpha, const float* __restrict__ m,
                               const float* __restrict__ h,
                               float* __restrict__ acc, float* __restrict__ denom,
                               int E_, int N_) {
    long long gid = (long long)blockIdx.x * blockDim.x + threadIdx.x;
    int i    = (int)(gid >> 5);
    int lane = (int)(gid & 31);
    if (i >= E_ + N_) return;
    int s, d;
    if (i < E_) { s = src[i]; d = dst[i]; }
    else        { s = d = i - E_; }
    float p = __expf(alpha[i] - m[d]);
    if (lane == 0) atomicAdd(denom + d, p);
    const float* hs = h + (long long)s * 64;
    float* ad2 = acc + (long long)d * 64;
    atomicAdd(ad2 + lane,      p * hs[lane]);
    atomicAdd(ad2 + lane + 32, p * hs[lane + 32]);
}

// normalize + bias (+ optional relu)
__global__ void finalize_kernel(const float* __restrict__ acc, const float* __restrict__ denom,
                                const float* __restrict__ bias, float* __restrict__ out,
                                int N_, int do_relu) {
    long long gid = (long long)blockIdx.x * blockDim.x + threadIdx.x;
    if (gid >= (long long)N_ * 64) return;
    int n = (int)(gid >> 6);
    int c = (int)(gid & 63);
    float v = acc[gid] / (denom[n] + 1e-16f) + bias[c];
    if (do_relu) v = fmaxf(v, 0.f);
    out[gid] = v;
}

// ---------------------------------------------------------------------------
extern "C" void kernel_launch(void* const* d_in, const int* in_sizes, int n_in,
                              void* d_out, int out_size, void* d_ws, size_t ws_size,
                              hipStream_t stream) {
    const float* x    = (const float*)d_in[0];
    const int*   ei   = (const int*)  d_in[1];
    const float* ea   = (const float*)d_in[2];
    const float* W1   = (const float*)d_in[3];
    const float* We1  = (const float*)d_in[4];
    const float* as1v = (const float*)d_in[5];
    const float* ad1v = (const float*)d_in[6];
    const float* ate1 = (const float*)d_in[7];
    const float* b1   = (const float*)d_in[8];
    const float* W2   = (const float*)d_in[9];
    const float* We2  = (const float*)d_in[10];
    const float* as2v = (const float*)d_in[11];
    const float* ad2v = (const float*)d_in[12];
    const float* ate2 = (const float*)d_in[13];
    const float* b2   = (const float*)d_in[14];

    const int N = in_sizes[0] / IN_C;
    const int E = in_sizes[1] / 2;
    const int* src = ei;
    const int* dst = ei + E;

    // workspace layout (floats)
    float* ws = (float*)d_ws;
    long long o = 0;
    float* ae1  = ws + o; o += E;
    float* ae2  = ws + o; o += E;
    float* sae1 = ws + o; o += N;
    float* sae2 = ws + o; o += N;
    float* cnt  = ws + o; o += N;
    float* v1   = ws + o; o += EDGE_D;
    float* v2   = ws + o; o += EDGE_D;
    float* as_1 = ws + o; o += N;
    float* ad_1 = ws + o; o += N;
    float* as_2 = ws + o; o += N;
    float* ad_2 = ws + o; o += N;
    float* m    = ws + o; o += N;
    float* den  = ws + o; o += N;
    float* alpha= ws + o; o += (long long)E + N;
    float* acc  = ws + o; o += (long long)N * 64;
    float* h    = ws + o; o += (long long)N * 64;   // h1, then reused as h2
    float* hrel = ws + o; o += (long long)N * 64;
    _Float16* W1p = (_Float16*)(ws + o); o += (IN_C  / 32) * 4 * 32 * 16 / 2;  // f16 halves per float
    _Float16* W2p = (_Float16*)(ws + o); o += (HID_C / 32) * 4 * 32 * 16 / 2;

    const int T = 256;
    const long long EN = (long long)E + N;

    // -------- shared precompute --------
    ve_kernel<<<1, 64, 0, stream>>>(We1, ate1, We2, ate2, v1, v2);
    pack_w_kernel<IN_C ><<<2, 256, 0, stream>>>(W1, W1p);   // 512 frags
    pack_w_kernel<HID_C><<<1, 256, 0, stream>>>(W2, W2p);   // 256 frags
    fill_kernel<<<cdiv(3LL * N, T), T, 0, stream>>>(sae1, 0.f, 3LL * N); // sae1,sae2,cnt contiguous
    edge_ae_kernel<<<cdiv(E, T), T, 0, stream>>>(ea, dst, v1, v2, ae1, ae2, sae1, sae2, cnt, E);

    const int gemm_blocks = cdiv(cdiv(N, 16), 8);   // 8 waves/block, 1 tile/wave

    // -------- layer 1 --------
    gemm_wmma<IN_C><<<gemm_blocks, T, 0, stream>>>(x, W1p, h, N);
    node_attn_kernel<<<cdiv(N, T), T, 0, stream>>>(h, as1v, ad1v, as_1, ad_1, N);
    fill_kernel<<<cdiv(N, T), T, 0, stream>>>(m, -INFINITY, N);
    fill_kernel<<<cdiv(N, T), T, 0, stream>>>(den, 0.f, N);
    fill_kernel<<<cdiv((long long)N * 64, T), T, 0, stream>>>(acc, 0.f, (long long)N * 64);
    alpha_max_kernel<<<cdiv(EN, T), T, 0, stream>>>(src, dst, ae1, sae1, cnt, as_1, ad_1, alpha, m, E, N);
    scatter_kernel<<<cdiv(EN * 32, T), T, 0, stream>>>(src, dst, alpha, m, h, acc, den, E, N);
    finalize_kernel<<<cdiv((long long)N * 64, T), T, 0, stream>>>(acc, den, b1, hrel, N, 1);

    // -------- layer 2 --------
    gemm_wmma<HID_C><<<gemm_blocks, T, 0, stream>>>(hrel, W2p, h, N);
    node_attn_kernel<<<cdiv(N, T), T, 0, stream>>>(h, as2v, ad2v, as_2, ad_2, N);
    fill_kernel<<<cdiv(N, T), T, 0, stream>>>(m, -INFINITY, N);
    fill_kernel<<<cdiv(N, T), T, 0, stream>>>(den, 0.f, N);
    fill_kernel<<<cdiv((long long)N * 64, T), T, 0, stream>>>(acc, 0.f, (long long)N * 64);
    alpha_max_kernel<<<cdiv(EN, T), T, 0, stream>>>(src, dst, ae2, sae2, cnt, as_2, ad_2, alpha, m, E, N);
    scatter_kernel<<<cdiv(EN * 32, T), T, 0, stream>>>(src, dst, alpha, m, h, acc, den, E, N);
    finalize_kernel<<<cdiv((long long)N * 64, T), T, 0, stream>>>(acc, den, b2, (float*)d_out, N, 0);
}
